// GravNetBlock_85864986182012
// MI455X (gfx1250) — matrix operand order repaired and checked by reading, not compiled, not run on hardware
//
#include <hip/hip_runtime.h>
#include <hip/hip_bf16.h>
#include <stdint.h>

typedef __attribute__((ext_vector_type(16))) _Float16 v16h;
typedef __attribute__((ext_vector_type(8)))  float    v8f;

#define NTOT 32768
#define NPB  4096
#define DD   64
#define KK   40
#define EPSV 1e-5f

__device__ __forceinline__ float eluf(float v){ return v > 0.f ? v : (__expf(v) - 1.f); }

// A fragment: 16x32 f16 tile from fp32 row-major [16 x ld], columns k0..k0+31.
// ISA layout: lanes 0-15 -> M=lane, K in {0..7,16..23}; lanes 16-31 -> +8 on K.
__device__ __forceinline__ v16h load_a(const float* base, int ld, int k0, int lane){
  int m   = lane & 15;
  int khi = (lane >> 4) << 3;
  const float* p = base + (size_t)m * ld + k0 + khi;
  v16h a;
#pragma unroll
  for (int j = 0; j < 16; ++j){
    int k = (j & 7) + ((j >> 3) << 4);
    a[j] = (_Float16)p[k];
  }
  return a;
}

// B fragment: 32x16 f16 tile from fp32 row-major W[K x ldb], rows k0.., cols n0..n0+15.
__device__ __forceinline__ v16h load_b(const float* W, int ldb, int k0, int n0, int lane){
  int n   = lane & 15;
  int khi = (lane >> 4) << 3;
  const float* p = W + (size_t)(k0 + khi) * ldb + n0 + n;
  v16h b;
#pragma unroll
  for (int j = 0; j < 16; ++j) b[j] = (_Float16)p[(size_t)((j & 7) + ((j >> 3) << 4)) * ldb];
  return b;
}

// ---------------- kernel 1: h = elu(elu(x@w1+b1)@w2+b2), accumulate BN1 stats ----
__global__ __launch_bounds__(256) void k_mlp(const float* __restrict__ x,
    const float* __restrict__ w1, const float* __restrict__ b1,
    const float* __restrict__ w2, const float* __restrict__ b2,
    float* __restrict__ h_out, float* __restrict__ stats){
  __shared__ float lt[8][16][68];
  __shared__ float ssum[64], ssq[64];
  int tid = threadIdx.x, wave = tid >> 5, lane = tid & 31;
  if (tid < 64){ ssum[tid] = 0.f; ssq[tid] = 0.f; }
  int row0 = blockIdx.x * 128 + wave * 16;

  v8f acc[4] = {};
#pragma unroll
  for (int k0 = 0; k0 < 64; k0 += 32){
    v16h a = load_a(x + (size_t)row0 * DD, DD, k0, lane);
#pragma unroll
    for (int nt = 0; nt < 4; ++nt){
      v16h b = load_b(w1, DD, k0, nt * 16, lane);
      acc[nt] = __builtin_amdgcn_wmma_f32_16x16x32_f16(false, a, false, b, (short)0, acc[nt], false, false);
    }
  }
  int n = lane & 15, mb = (lane >> 4) << 3;
#pragma unroll
  for (int nt = 0; nt < 4; ++nt){
    float bia = b1[nt * 16 + n];
#pragma unroll
    for (int r = 0; r < 8; ++r)
      lt[wave][mb + r][nt * 16 + n] = eluf(acc[nt][r] + bia);
  }
  __syncthreads();

  v8f acc2[4] = {};
#pragma unroll
  for (int k0 = 0; k0 < 64; k0 += 32){
    v16h a = load_a(&lt[wave][0][0], 68, k0, lane);
#pragma unroll
    for (int nt = 0; nt < 4; ++nt){
      v16h b = load_b(w2, DD, k0, nt * 16, lane);
      acc2[nt] = __builtin_amdgcn_wmma_f32_16x16x32_f16(false, a, false, b, (short)0, acc2[nt], false, false);
    }
  }
#pragma unroll
  for (int nt = 0; nt < 4; ++nt){
    float bia = b2[nt * 16 + n];
    float ps = 0.f, pq = 0.f;
#pragma unroll
    for (int r = 0; r < 8; ++r){
      float v = eluf(acc2[nt][r] + bia);
      h_out[(size_t)(row0 + mb + r) * DD + nt * 16 + n] = v;
      ps += v; pq += v * v;
    }
    atomicAdd(&ssum[nt * 16 + n], ps);
    atomicAdd(&ssq [nt * 16 + n], pq);
  }
  __syncthreads();
  if (tid < 64){
    atomicAdd(&stats[tid],      ssum[tid]);
    atomicAdd(&stats[64 + tid], ssq[tid]);
  }
}

// ---------------- kernel 2: hbn = BN1(h); f = hbn@wh+bh; s = hbn@ws+bs ----------
__global__ __launch_bounds__(256) void k_bnsf(const float* __restrict__ h,
    const float* __restrict__ stats,
    const float* __restrict__ gamma, const float* __restrict__ beta,
    const float* __restrict__ wsw, const float* __restrict__ bsv,
    const float* __restrict__ wh,  const float* __restrict__ bh,
    float* __restrict__ hbn, float* __restrict__ sarr, float* __restrict__ farr){
  __shared__ float lt[8][16][68];
  __shared__ float sA[64], sC[64];
  int tid = threadIdx.x, wave = tid >> 5, lane = tid & 31;
  if (tid < 64){
    float m   = stats[tid] * (1.f / NTOT);
    float v   = stats[64 + tid] * (1.f / NTOT) - m * m;
    float inv = rsqrtf(v + EPSV);
    float a   = gamma[tid] * inv;
    sA[tid] = a; sC[tid] = beta[tid] - m * a;
  }
  __syncthreads();
  int row0 = blockIdx.x * 128 + wave * 16;

  for (int t = lane; t < 16 * 64; t += 32){
    int r = t >> 6, c = t & 63;
    float v = h[(size_t)(row0 + r) * DD + c];
    v = v * sA[c] + sC[c];
    lt[wave][r][c] = v;
    hbn[(size_t)(row0 + r) * DD + c] = v;
  }

  v8f acc[4] = {};
#pragma unroll
  for (int k0 = 0; k0 < 64; k0 += 32){
    v16h a = load_a(&lt[wave][0][0], 68, k0, lane);
#pragma unroll
    for (int nt = 0; nt < 4; ++nt){
      v16h b = load_b(wh, DD, k0, nt * 16, lane);
      acc[nt] = __builtin_amdgcn_wmma_f32_16x16x32_f16(false, a, false, b, (short)0, acc[nt], false, false);
    }
  }
  int n = lane & 15, mb = (lane >> 4) << 3;
#pragma unroll
  for (int nt = 0; nt < 4; ++nt){
    float bia = bh[nt * 16 + n];
#pragma unroll
    for (int r = 0; r < 8; ++r)
      farr[(size_t)(row0 + mb + r) * DD + nt * 16 + n] = acc[nt][r] + bia;
  }

  if (lane < 16){
    int r = lane;
    float a0 = bsv[0], a1 = bsv[1], a2 = bsv[2];
#pragma unroll 8
    for (int k = 0; k < 64; ++k){
      float hv = lt[wave][r][k];
      a0 += hv * wsw[k * 3 + 0];
      a1 += hv * wsw[k * 3 + 1];
      a2 += hv * wsw[k * 3 + 2];
    }
    size_t o = (size_t)(row0 + r) * 3;
    sarr[o] = a0; sarr[o + 1] = a1; sarr[o + 2] = a2;
  }
}

// ---------------- kernel 3: per-batch KNN (top-40 by d2) -> idx, w=exp(-10*d2) ---
// Candidate tiles double-buffered in LDS via async global->LDS DMA (ASYNCcnt).
#define TS 512
__global__ __launch_bounds__(128) void k_knn(const float* __restrict__ sarr,
    int* __restrict__ idx, float* __restrict__ wgt){
  __shared__ __align__(16) float cs2[2][TS * 3];
  __shared__ float kd[128 * KK];
  __shared__ int   ki[128 * KK];
  int tid = threadIdx.x;
  int b = blockIdx.x >> 5;
  int chunk = blockIdx.x & 31;
  int i = b * NPB + chunk * 128 + tid;
  float qx = sarr[(size_t)i * 3 + 0];
  float qy = sarr[(size_t)i * 3 + 1];
  float qz = sarr[(size_t)i * 3 + 2];
  float* myd = kd + tid * KK;
  int*   myi = ki + tid * KK;
#pragma unroll
  for (int k = 0; k < KK; ++k){ myd[k] = 3.4e38f; myi[k] = i; }
  float worst = 3.4e38f; int wpos = 0;

  const float* batch_s = sarr + (size_t)b * NPB * 3;

  // prime buffer 0: 3 x b128 async loads per thread (1536 floats total)
#pragma unroll
  for (int u = 0; u < 3; ++u){
    int e4 = tid + u * 128;
    uint32_t lds = (uint32_t)(uintptr_t)&cs2[0][e4 * 4];
    const float* g = batch_s + (size_t)e4 * 4;
    asm volatile("global_load_async_to_lds_b128 %0, %1, off"
                 :: "v"(lds), "v"(g) : "memory");
  }

  for (int t0 = 0; t0 < NPB; t0 += TS){
    int buf = (t0 / TS) & 1;
    asm volatile("s_wait_asynccnt 0" ::: "memory");
    __syncthreads();                       // tile `buf` resident for all threads
    if (t0 + TS < NPB){                    // issue DMA for next tile into other buffer
#pragma unroll
      for (int u = 0; u < 3; ++u){
        int e4 = tid + u * 128;
        uint32_t lds = (uint32_t)(uintptr_t)&cs2[buf ^ 1][e4 * 4];
        const float* g = batch_s + (size_t)(t0 + TS) * 3 + (size_t)e4 * 4;
        asm volatile("global_load_async_to_lds_b128 %0, %1, off"
                     :: "v"(lds), "v"(g) : "memory");
      }
    }
    const float* cs = cs2[buf];
    for (int c = 0; c < TS; ++c){
      float dx = qx - cs[c * 3 + 0];
      float dy = qy - cs[c * 3 + 1];
      float dz = qz - cs[c * 3 + 2];
      float d2 = dx * dx + dy * dy + dz * dz;
      if (d2 < worst){
        myd[wpos] = d2; myi[wpos] = b * NPB + t0 + c;
        float w = myd[0]; int p = 0;
#pragma unroll
        for (int k = 1; k < KK; ++k){ float v = myd[k]; if (v > w){ w = v; p = k; } }
        worst = w; wpos = p;
      }
    }
    __syncthreads();                       // all reads of `buf` done before reuse
  }
  size_t o = (size_t)i * KK;
#pragma unroll
  for (int k = 0; k < KK; ++k){
    idx[o + k] = myi[k];
    wgt[o + k] = __expf(-10.f * myd[k]);
  }
}

// ---------------- kernel 4: neighbor aggregation (mean/max over K) ---------------
__global__ __launch_bounds__(256) void k_agg(const float* __restrict__ farr,
    const int* __restrict__ idx, const float* __restrict__ wgt,
    float* __restrict__ aggm, float* __restrict__ aggx){
  int wave = threadIdx.x >> 5, lane = threadIdx.x & 31;
  int i = blockIdx.x * 8 + wave;
  const int*   ip = idx + (size_t)i * KK;
  const float* wp = wgt + (size_t)i * KK;
  float m0 = 0.f, m1 = 0.f, x0 = -3.4e38f, x1 = -3.4e38f;
  int jn = ip[0];
  for (int k = 0; k < KK; ++k){
    int j = jn;
    float wv = wp[k];
    if (k + 1 < KK){
      jn = ip[k + 1];
      __builtin_prefetch(farr + (size_t)jn * DD + lane, 0, 1);  // global_prefetch_b8
    }
    float v0 = farr[(size_t)j * DD + lane]      * wv;
    float v1 = farr[(size_t)j * DD + lane + 32] * wv;
    m0 += v0; m1 += v1;
    x0 = fmaxf(x0, v0); x1 = fmaxf(x1, v1);
  }
  const float s = 1.f / KK;
  aggm[(size_t)i * DD + lane]      = m0 * s;
  aggm[(size_t)i * DD + lane + 32] = m1 * s;
  aggx[(size_t)i * DD + lane]      = x0;
  aggx[(size_t)i * DD + lane + 32] = x1;
}

// ---------------- kernel 5: out_pre = [mean|max|hbn]@wo+bo; accumulate BN2 stats -
__global__ __launch_bounds__(256) void k_out(const float* __restrict__ aggm,
    const float* __restrict__ aggx, const float* __restrict__ hbn,
    const float* __restrict__ wo, const float* __restrict__ bo,
    float* __restrict__ pre, float* __restrict__ stats){
  __shared__ float ssum[64], ssq[64];
  int tid = threadIdx.x, wave = tid >> 5, lane = tid & 31;
  if (tid < 64){ ssum[tid] = 0.f; ssq[tid] = 0.f; }
  __syncthreads();
  int row0 = blockIdx.x * 128 + wave * 16;
  v8f acc[4] = {};
  const float* srcs[3] = { aggm, aggx, hbn };
#pragma unroll
  for (int k0 = 0; k0 < 192; k0 += 32){
    const float* src = srcs[k0 >> 6];
    v16h a = load_a(src + (size_t)row0 * DD, DD, k0 & 63, lane);
#pragma unroll
    for (int nt = 0; nt < 4; ++nt){
      v16h b = load_b(wo, DD, k0, nt * 16, lane);
      acc[nt] = __builtin_amdgcn_wmma_f32_16x16x32_f16(false, a, false, b, (short)0, acc[nt], false, false);
    }
  }
  int n = lane & 15, mb = (lane >> 4) << 3;
#pragma unroll
  for (int nt = 0; nt < 4; ++nt){
    float bia = bo[nt * 16 + n];
    float ps = 0.f, pq = 0.f;
#pragma unroll
    for (int r = 0; r < 8; ++r){
      float v = acc[nt][r] + bia;
      pre[(size_t)(row0 + mb + r) * DD + nt * 16 + n] = v;
      ps += v; pq += v * v;
    }
    atomicAdd(&ssum[nt * 16 + n], ps);
    atomicAdd(&ssq [nt * 16 + n], pq);
  }
  __syncthreads();
  if (tid < 64){
    atomicAdd(&stats[tid],      ssum[tid]);
    atomicAdd(&stats[64 + tid], ssq[tid]);
  }
}

// ---------------- kernel 6: apply BN2 -> d_out ----------------------------------
__global__ __launch_bounds__(256) void k_bn2(const float* __restrict__ pre,
    const float* __restrict__ stats,
    const float* __restrict__ gamma, const float* __restrict__ beta,
    float* __restrict__ out){
  int t = blockIdx.x * 256 + threadIdx.x;
  int c = t & 63;
  float m   = stats[c] * (1.f / NTOT);
  float v   = stats[64 + c] * (1.f / NTOT) - m * m;
  float inv = rsqrtf(v + EPSV);
  out[t] = (pre[t] - m) * inv * gamma[c] + beta[c];
}

__global__ void k_zero(float* __restrict__ p){ p[threadIdx.x] = 0.f; }

extern "C" void kernel_launch(void* const* d_in, const int* in_sizes, int n_in,
                              void* d_out, int out_size, void* d_ws, size_t ws_size,
                              hipStream_t stream){
  (void)in_sizes; (void)n_in; (void)out_size; (void)ws_size;
  const float* x      = (const float*)d_in[0];
  // d_in[1] = batch (int64) — unused: batches are contiguous NPB-sized slabs
  const float* w1     = (const float*)d_in[2];
  const float* b1     = (const float*)d_in[3];
  const float* w2     = (const float*)d_in[4];
  const float* b2     = (const float*)d_in[5];
  const float* gamma1 = (const float*)d_in[6];
  const float* beta1  = (const float*)d_in[7];
  const float* wsw    = (const float*)d_in[8];
  const float* bsv    = (const float*)d_in[9];
  const float* wh     = (const float*)d_in[10];
  const float* bh     = (const float*)d_in[11];
  const float* wo     = (const float*)d_in[12];
  const float* bo     = (const float*)d_in[13];
  const float* gamma2 = (const float*)d_in[14];
  const float* beta2  = (const float*)d_in[15];

  float* base = (float*)d_ws;
  const size_t NF = (size_t)NTOT * DD;
  float* h     = base;
  float* hbn   = h    + NF;
  float* farr  = hbn  + NF;
  float* sarr  = farr + NF;
  float* aggm  = sarr + (size_t)NTOT * 3;
  float* aggx  = aggm + NF;
  float* pre   = aggx + NF;
  float* wgt   = pre  + NF;
  int*   idx   = (int*)(wgt + (size_t)NTOT * KK);
  float* stats = (float*)(idx + (size_t)NTOT * KK);   // [0..127]=BN1, [128..255]=BN2

  k_zero<<<1, 256, 0, stream>>>(stats);
  k_mlp <<<256, 256, 0, stream>>>(x, w1, b1, w2, b2, h, stats);
  k_bnsf<<<256, 256, 0, stream>>>(h, stats, gamma1, beta1, wsw, bsv, wh, bh, hbn, sarr, farr);
  k_knn <<<256, 128, 0, stream>>>(sarr, idx, wgt);
  k_agg <<<4096, 256, 0, stream>>>(farr, idx, wgt, aggm, aggx);
  k_out <<<256, 256, 0, stream>>>(aggm, aggx, hbn, wo, bo, pre, stats + 128);
  k_bn2 <<<NTOT * DD / 256, 256, 0, stream>>>(pre, stats + 128, gamma2, beta2, (float*)d_out);
}
